// scaled_dot_product_attention_2327872274492
// MI455X (gfx1250) — matrix operand order repaired and checked by reading, not compiled, not run on hardware
//
#include <hip/hip_runtime.h>

#define B_    16
#define L_    2048
#define D_    128
#define SCALE 0.08838834764831845f   // 1/sqrt(128)

typedef __attribute__((ext_vector_type(16))) __bf16 v16bf;
typedef __attribute__((ext_vector_type(8)))  float  v8f;

// Load 8 consecutive fp32 from p, convert to bf16, place at elements [e0 .. e0+7].
__device__ __forceinline__ void cvt8(const float* __restrict__ p, v16bf& a, int e0) {
    float4 f0 = *(const float4*)p;
    float4 f1 = *(const float4*)(p + 4);
    a[e0 + 0] = (__bf16)f0.x; a[e0 + 1] = (__bf16)f0.y;
    a[e0 + 2] = (__bf16)f0.z; a[e0 + 3] = (__bf16)f0.w;
    a[e0 + 4] = (__bf16)f1.x; a[e0 + 5] = (__bf16)f1.y;
    a[e0 + 6] = (__bf16)f1.z; a[e0 + 7] = (__bf16)f1.w;
}

// Load 16 consecutive fp32 from p, convert to bf16 elements [0..15].
__device__ __forceinline__ void cvt16(const float* __restrict__ p, v16bf& a) {
    cvt8(p, a, 0);
    cvt8(p + 8, a, 8);
}

extern "C" __global__ void __launch_bounds__(256)
sdpa_wmma_kernel(const float* __restrict__ q, const float* __restrict__ k,
                 const float* __restrict__ v, float* __restrict__ ctx,
                 float* __restrict__ attn) {
    extern __shared__ float smem[];          // 16 rows x 2048 cols fp32 = 128 KB

    const int bid  = blockIdx.x;
    const int b    = bid >> 7;               // 128 query tiles per batch
    const int q0   = (bid & 127) << 4;       // first query row of this tile
    const int tid  = threadIdx.x;
    const int wv   = tid >> 5;               // wave id 0..7
    const int lane = tid & 31;
    const int hl   = lane >> 4;              // half-wave select
    const int mn   = lane & 15;              // row (A/C) or column (B) index

    // ---------------- Phase 1: S = (Q K^T) * scale -> LDS ----------------
    // Hoist Q tile into 4 A-fragments (ISA 16-bit A 16x32 layout):
    // lane holds row M=mn; element e: K = (e>>3)*16 + hl*8 + (e&7).
    const float* qrow = q + ((size_t)(b * L_ + q0 + mn)) * D_;
    v16bf A0, A1, A2, A3;
    cvt8(qrow +  0 + hl * 8, A0, 0);  cvt8(qrow + 16 + hl * 8, A0, 8);
    cvt8(qrow + 32 + hl * 8, A1, 0);  cvt8(qrow + 48 + hl * 8, A1, 8);
    cvt8(qrow + 64 + hl * 8, A2, 0);  cvt8(qrow + 80 + hl * 8, A2, 8);
    cvt8(qrow + 96 + hl * 8, A3, 0);  cvt8(qrow + 112 + hl * 8, A3, 8);

    // Each wave computes 16 consecutive 16x16 score tiles.
#pragma unroll 1
    for (int jj = 0; jj < 16; ++jj) {
        const int j  = wv * 16 + jj;         // key tile index 0..127
        const int kb = j << 4;               // first key of tile
        // B 32x16 layout: lane holds column N=mn (key kb+mn);
        // element e: K-dim feature d = chunk*32 + (e>>3)*16 + hl*8 + (e&7).
        const float* krow = k + ((size_t)(b * L_ + kb + mn)) * D_;

        v8f acc = {0.f, 0.f, 0.f, 0.f, 0.f, 0.f, 0.f, 0.f};
        v16bf Bt;
        cvt8(krow +  0 + hl * 8, Bt, 0);  cvt8(krow + 16 + hl * 8, Bt, 8);
        acc = __builtin_amdgcn_wmma_f32_16x16x32_bf16(false, A0, false, Bt, (short)0, acc, false, false);
        cvt8(krow + 32 + hl * 8, Bt, 0);  cvt8(krow + 48 + hl * 8, Bt, 8);
        acc = __builtin_amdgcn_wmma_f32_16x16x32_bf16(false, A1, false, Bt, (short)0, acc, false, false);
        cvt8(krow + 64 + hl * 8, Bt, 0);  cvt8(krow + 80 + hl * 8, Bt, 8);
        acc = __builtin_amdgcn_wmma_f32_16x16x32_bf16(false, A2, false, Bt, (short)0, acc, false, false);
        cvt8(krow + 96 + hl * 8, Bt, 0);  cvt8(krow + 112 + hl * 8, Bt, 8);
        acc = __builtin_amdgcn_wmma_f32_16x16x32_bf16(false, A3, false, Bt, (short)0, acc, false, false);

        // C layout: VGPR i -> row i + 8*hl, column mn.
#pragma unroll
        for (int i = 0; i < 8; ++i)
            smem[(i + hl * 8) * L_ + kb + mn] = acc[i] * SCALE;
    }
    __syncthreads();

    // ---------------- Phase 2: row softmax (fp32), write attention -------
    // Wave w owns rows 2w and 2w+1; 32 lanes stride the 2048 columns.
#pragma unroll 1
    for (int rr = 0; rr < 2; ++rr) {
        const int row = wv * 2 + rr;
        float* srow = smem + row * L_;

        float mx = -3.402823466e38f;
        for (int c = lane; c < L_; c += 32) mx = fmaxf(mx, srow[c]);
#pragma unroll
        for (int off = 16; off > 0; off >>= 1) mx = fmaxf(mx, __shfl_xor(mx, off, 32));

        float sum = 0.f;
        for (int c = lane; c < L_; c += 32) {
            float e = __expf(srow[c] - mx);
            srow[c] = e;
            sum += e;
        }
#pragma unroll
        for (int off = 16; off > 0; off >>= 1) sum += __shfl_xor(sum, off, 32);
        const float inv = 1.f / sum;

        // Attention matrix is write-once streaming output (256 MB total):
        // non-temporal stores keep the 192 MB L2 for Q/K/V reuse.
        float* arow = attn + ((size_t)(b * L_ + q0 + row)) * L_;
        for (int c = lane; c < L_; c += 32) {
            float p = srow[c] * inv;
            srow[c] = p;                          // keep probs in LDS for PV
            __builtin_nontemporal_store(p, arow + c);
        }
    }
    __syncthreads();

    // ---------------- Phase 3: context = P @ V ---------------------------
    // Wave w owns output feature slice d0 = 16*w; accumulate over 64 K=32 chunks.
    const int d0 = wv * 16;
    v8f cacc = {0.f, 0.f, 0.f, 0.f, 0.f, 0.f, 0.f, 0.f};
#pragma unroll 1
    for (int j = 0; j < 64; ++j) {
        // A from LDS probs: row M=mn, K(key) = j*32 + (e>>3)*16 + hl*8 + (e&7).
        const float* prow = smem + mn * L_ + j * 32;
        v16bf Ap;
        cvt8(prow + hl * 8, Ap, 0);
        cvt8(prow + 16 + hl * 8, Ap, 8);

        // B from V: column N=mn -> feature d0+mn; element e -> key j*32 + (e>>3)*16 + hl*8 + (e&7).
        const float* vcol = v + ((size_t)(b * L_ + j * 32 + hl * 8)) * D_ + d0 + mn;
        v16bf Bv;
#pragma unroll
        for (int e = 0; e < 16; ++e) Bv[e] = (__bf16)vcol[((e >> 3) * 16 + (e & 7)) * D_];

        cacc = __builtin_amdgcn_wmma_f32_16x16x32_bf16(false, Ap, false, Bv, (short)0, cacc, false, false);
    }

    // C layout: VGPR i -> query row i + 8*hl, column d0+mn.
    // Context is also write-once: stream it past L2 with NT stores.
    float* crow = ctx + ((size_t)(b * L_ + q0)) * D_ + d0 + mn;
#pragma unroll
    for (int i = 0; i < 8; ++i)
        __builtin_nontemporal_store(cacc[i], crow + (size_t)(i + hl * 8) * D_);
}

extern "C" void kernel_launch(void* const* d_in, const int* in_sizes, int n_in,
                              void* d_out, int out_size, void* d_ws, size_t ws_size,
                              hipStream_t stream) {
    const float* q = (const float*)d_in[0];
    const float* k = (const float*)d_in[1];
    const float* v = (const float*)d_in[2];
    float* ctx  = (float*)d_out;
    float* attn = ctx + (size_t)B_ * L_ * D_;   // outputs concatenated: context, attention

    dim3 grid(B_ * (L_ / 16));                  // 2048 workgroups
    dim3 block(256);                            // 8 wave32 waves
    size_t lds_bytes = (size_t)16 * L_ * sizeof(float);  // 128 KB score/prob strip

    sdpa_wmma_kernel<<<grid, block, lds_bytes, stream>>>(q, k, v, ctx, attn);
}
